// NeuralODE_14053132992739
// MI455X (gfx1250) — compile-verified, matrix operand order checked by reading, not run on hardware
//
#include <hip/hip_runtime.h>
#include <math.h>

typedef __attribute__((ext_vector_type(16))) _Float16 v16h;
typedef __attribute__((ext_vector_type(8)))  float    v8f;
typedef __attribute__((ext_vector_type(4)))  float    v4f;

#define STATE_DIM 64
#define HIDDEN    256
#define BATCH     4096
#define MT        16      // batch rows per workgroup
#define THREADS   256     // 8 wave32 waves

// LDS layout (offsets in _Float16 units for the f16 region)
#define OFF_W1F 0         // 32 blocks  * 512 halves  (2 ktiles x 16 ntiles)
#define OFF_W2F 16384     // 128 blocks * 512 halves  (8 x 16)
#define OFF_W3F 81920     // 32 blocks  * 512 halves  (8 x 4)
#define OFF_H1F 98304     // 8 blocks (16x256 as A-fragments)
#define OFF_H2F 102400    // 8 blocks
#define OFF_XF  106496    // 2 blocks (16x64 as A-fragments)
#define BYTE_F32 215040   // start of f32 region (byte offset, 16-aligned)
#define LDS_BYTES 221440

// ---- hardware tanh (CDNA5 TRANS32 op, v_tanh_f32) --------------------------
__device__ __forceinline__ float fast_tanh(float x) {
#if __has_builtin(__builtin_amdgcn_tanhf)
    return __builtin_amdgcn_tanhf(x);
#elif __has_builtin(__builtin_amdgcn_tanh_f32)
    return __builtin_amdgcn_tanh_f32(x);
#else
    float r;
    asm volatile("v_tanh_f32 %0, %1" : "=v"(r) : "v"(x));
    return r;
#endif
}

// K-index within a 32-wide k-tile for fragment element e (0..15), lane group g (0..1)
// per CDNA5 ISA 16-bit A/B fragment layout.
__device__ __forceinline__ int kmap(int e, int g) {
    return e + 8 * g + ((e >= 8) ? 8 : 0);
}
// inverse: K32 (0..31) -> (element e, lane-group g)
__device__ __forceinline__ void kinv(int K, int& e, int& g) {
    g = (K >> 3) & 1;
    e = K - 8 * g - ((K >= 16) ? 8 : 0);
}

__device__ __forceinline__ v8f wmma_f16(v16h a, v16h b, v8f c) {
    return __builtin_amdgcn_wmma_f32_16x16x32_f16(
        /*neg_a=*/false, a, /*neg_b=*/false, b,
        /*c_mod=*/(short)0, c, /*reuse_a=*/false, /*reuse_b=*/false);
}

// Pack an f32 row-major [K x ncols] weight matrix into f16 WMMA B-fragment blocks:
// block(kt,nt) = 32 lanes x 16 halves, contiguous 32B per lane.
__device__ __forceinline__ void swizzle_weights(const float* __restrict__ W,
                                                _Float16* dst, int NT, int nblocks,
                                                int ncols) {
    int total = nblocks * 512;
    for (int i = threadIdx.x; i < total; i += THREADS) {
        int blk  = i >> 9;
        int rem  = i & 511;
        int lane = rem >> 4;
        int e    = rem & 15;
        int kt = blk / NT, nt = blk % NT;
        int g = lane >> 4;
        int k = kt * 32 + kmap(e, g);
        int n = nt * 16 + (lane & 15);
        dst[i] = (_Float16)W[(size_t)k * ncols + n];
    }
}

__global__ void __launch_bounds__(THREADS)
node_persistent_kernel(const float* __restrict__ x0,
                       const float* __restrict__ W1, const float* __restrict__ b1,
                       const float* __restrict__ W2, const float* __restrict__ b2,
                       const float* __restrict__ W3, const float* __restrict__ b3,
                       const float* __restrict__ dt_scale,
                       const int*   __restrict__ steps_p,
                       float* __restrict__ out) {
    extern __shared__ char smem_raw[];
    _Float16* hb  = (_Float16*)smem_raw;
    _Float16* w1f = hb + OFF_W1F;
    _Float16* w2f = hb + OFF_W2F;
    _Float16* w3f = hb + OFF_W3F;
    _Float16* h1f = hb + OFF_H1F;
    _Float16* h2f = hb + OFF_H2F;
    _Float16* xf  = hb + OFF_XF;
    float* f32b = (float*)(smem_raw + BYTE_F32);
    float* x32  = f32b;          // 16 x 64 state in f32 (Euler accumulation precision)
    float* b1s  = f32b + 1024;   // 256
    float* b2s  = f32b + 1280;   // 256
    float* b3s  = f32b + 1536;   // 64

    const int tid   = threadIdx.x;
    const int steps = steps_p[0];
    const int T     = steps + 1;
    const float scale = dt_scale[0] * 0.01f;   // dt_scale * DT
    const int r0 = blockIdx.x * MT;

    // ---- one-time setup: weights -> LDS (f16, fragment-swizzled), biases -> LDS ----
    swizzle_weights(W1, w1f, 16, 32,  HIDDEN);
    swizzle_weights(W2, w2f, 16, 128, HIDDEN);
    swizzle_weights(W3, w3f, 4,  32,  STATE_DIM);
    for (int i = tid; i < HIDDEN; i += THREADS) { b1s[i] = b1[i]; b2s[i] = b2[i]; }
    for (int i = tid; i < STATE_DIM; i += THREADS) b3s[i] = b3[i];

    // ---- load x0 tile, emit trajectory t=0 ----
    for (int i = tid; i < MT * STATE_DIM; i += THREADS) {
        int m = i >> 6, s = i & 63;
        float v = x0[(size_t)(r0 + m) * STATE_DIM + s];
        x32[i] = v;
        out[((size_t)(r0 + m) * T) * STATE_DIM + s] = v;
    }
    __syncthreads();
    // build x A-fragments (2 k-tiles over STATE_DIM=64)
    for (int i = tid; i < 1024; i += THREADS) {
        int blk = i >> 9, rem = i & 511;
        int lane = rem >> 4, e = rem & 15;
        int g = lane >> 4;
        int k = blk * 32 + kmap(e, g);
        xf[i] = (_Float16)x32[(lane & 15) * STATE_DIM + k];
    }
    __syncthreads();

    const int w    = tid >> 5;   // wave id 0..7 (uniform per wave)
    const int lane = tid & 31;
    const int g    = lane >> 4;  // C/D lane group
    const int n    = lane & 15;  // C/D column within tile

    // cooperative trajectory-store mapping: 16 threads per batch row, b128 each
    const int sm = tid >> 4;        // row 0..15
    const int ss = (tid & 15) * 4;  // col 0..63 step 4

    for (int t = 0; t < steps; ++t) {
        // ======== layer 1: h1 = tanh(x @ W1 + b1)  (K=64, wave owns ntiles 2w,2w+1) ====
        {
            v8f c0 = {}, c1 = {};
#pragma unroll
            for (int kt = 0; kt < 2; ++kt) {
                v16h a  = *(const v16h*)(xf + kt * 512 + lane * 16);
                v16h bA = *(const v16h*)(w1f + (kt * 16 + 2 * w + 0) * 512 + lane * 16);
                v16h bB = *(const v16h*)(w1f + (kt * 16 + 2 * w + 1) * 512 + lane * 16);
                c0 = wmma_f16(a, bA, c0);
                c1 = wmma_f16(a, bB, c1);
            }
            _Float16* blk = h1f + w * 512;   // this wave produces layer-2 k-tile w
#pragma unroll
            for (int c = 0; c < 2; ++c) {
                v8f acc = c ? c1 : c0;
                float bias = b1s[(2 * w + c) * 16 + n];
                int e, ga; kinv(c * 16 + n, e, ga);
#pragma unroll
                for (int j = 0; j < 8; ++j) {
                    int M = j + 8 * g;
                    blk[(M + 16 * ga) * 16 + e] = (_Float16)fast_tanh(acc[j] + bias);
                }
            }
        }
        __syncthreads();
        // ======== layer 2: h2 = tanh(h1 @ W2 + b2)  (K=256) ========
        {
            v8f c0 = {}, c1 = {};
#pragma unroll
            for (int kt = 0; kt < 8; ++kt) {
                v16h a  = *(const v16h*)(h1f + kt * 512 + lane * 16);
                v16h bA = *(const v16h*)(w2f + (kt * 16 + 2 * w + 0) * 512 + lane * 16);
                v16h bB = *(const v16h*)(w2f + (kt * 16 + 2 * w + 1) * 512 + lane * 16);
                c0 = wmma_f16(a, bA, c0);
                c1 = wmma_f16(a, bB, c1);
            }
            _Float16* blk = h2f + w * 512;
#pragma unroll
            for (int c = 0; c < 2; ++c) {
                v8f acc = c ? c1 : c0;
                float bias = b2s[(2 * w + c) * 16 + n];
                int e, ga; kinv(c * 16 + n, e, ga);
#pragma unroll
                for (int j = 0; j < 8; ++j) {
                    int M = j + 8 * g;
                    blk[(M + 16 * ga) * 16 + e] = (_Float16)fast_tanh(acc[j] + bias);
                }
            }
        }
        __syncthreads();
        // ======== layer 3: x += (h2 @ W3 + b3) * scale (state update in LDS) ========
        if (w < 4) {                       // wave-uniform branch -> EXEC all-ones
            v8f c3 = {};
#pragma unroll
            for (int kt = 0; kt < 8; ++kt) {
                v16h a  = *(const v16h*)(h2f + kt * 512 + lane * 16);
                v16h bb = *(const v16h*)(w3f + (kt * 4 + w) * 512 + lane * 16);
                c3 = wmma_f16(a, bb, c3);
            }
            int col = w * 16 + n;
            float bias = b3s[col];
            int e, ga; kinv((w & 1) * 16 + n, e, ga);
            _Float16* xblk = xf + (w >> 1) * 512;
#pragma unroll
            for (int j = 0; j < 8; ++j) {
                int M = j + 8 * g;
                float xn = x32[M * STATE_DIM + col] + (c3[j] + bias) * scale;
                x32[M * STATE_DIM + col] = xn;
                xblk[(M + 16 * ga) * 16 + e] = (_Float16)xn;
            }
        }
        __syncthreads();
        // ======== cooperative coalesced trajectory store: 256 x b128 = 4KB tile ====
        {
            v4f v = *(const v4f*)(x32 + sm * STATE_DIM + ss);
            *(v4f*)(out + ((size_t)(r0 + sm) * T + (t + 1)) * STATE_DIM + ss) = v;
        }
        // no barrier needed: x32 next written in layer 3 of the next step,
        // which is separated from these reads by that step's two barriers.
    }
}

extern "C" void kernel_launch(void* const* d_in, const int* in_sizes, int n_in,
                              void* d_out, int out_size, void* d_ws, size_t ws_size,
                              hipStream_t stream) {
    (void)in_sizes; (void)n_in; (void)out_size; (void)d_ws; (void)ws_size;
    const float* x0 = (const float*)d_in[0];
    const float* W1 = (const float*)d_in[1];
    const float* b1 = (const float*)d_in[2];
    const float* W2 = (const float*)d_in[3];
    const float* b2 = (const float*)d_in[4];
    const float* W3 = (const float*)d_in[5];
    const float* b3 = (const float*)d_in[6];
    const float* dt = (const float*)d_in[7];
    const int* steps = (const int*)d_in[8];
    float* out = (float*)d_out;

    dim3 grid(BATCH / MT);   // 256 workgroups
    dim3 block(THREADS);     // 8 wave32 waves
    hipLaunchKernelGGL(node_persistent_kernel, grid, block, LDS_BYTES, stream,
                       x0, W1, b1, W2, b2, W3, b3, dt, steps, out);
}
